// WindowAttention_13615046328490
// MI455X (gfx1250) — matrix operand order repaired and checked by reading, not compiled, not run on hardware
//
#include <hip/hip_runtime.h>

// ---------------- problem constants ----------------
#define BATCH   8
#define HH      256
#define WWID    256
#define CDIM    128
#define WS      8
#define NTOK    64          // tokens per window
#define NHD     4
#define HDIM    32
#define SHIFTSZ 4
#define NWH     32          // windows per side
#define NWIN    (BATCH*NWH*NWH)   // 8192
#define SCALEF  0.17677669529663687f

// ---------------- types ----------------
typedef __attribute__((ext_vector_type(16))) __bf16 v16bf;
typedef __attribute__((ext_vector_type(8)))  float  v8f;
typedef __attribute__((ext_vector_type(16))) unsigned short v16us;
typedef __attribute__((ext_vector_type(4)))  unsigned int  v4ui;
typedef __attribute__((ext_vector_type(4)))  int           v4i;

static __device__ __forceinline__ unsigned short f32_to_bf16(float f) {
  unsigned int u = __float_as_uint(f);
  u += 0x7FFFu + ((u >> 16) & 1u);      // round-to-nearest-even
  return (unsigned short)(u >> 16);
}

// pack two f32 -> {lo,hi} bf16 in one u32 (single VALU op when HW supports it)
#if __has_builtin(__builtin_amdgcn_cvt_pk_bf16_f32)
typedef __attribute__((ext_vector_type(2))) __bf16 v2bf;
static __device__ __forceinline__ unsigned int pack2_bf16(float lo, float hi) {
  v2bf v = __builtin_amdgcn_cvt_pk_bf16_f32(lo, hi);
  return __builtin_bit_cast(unsigned int, v);
}
#else
static __device__ __forceinline__ unsigned int pack2_bf16(float lo, float hi) {
  return (unsigned int)f32_to_bf16(lo) | ((unsigned int)f32_to_bf16(hi) << 16);
}
#endif

struct Frag32B { v4ui a, b; };

// A fragment: 16x32 bf16, M=16 rows, K=32.  contiguous K order chosen here (same as the B loaders):
// lane L holds row M=L%16; VGPR0-3: K = kh*16 + 0..7, VGPR4-7: K = kh*16 + 8..15
static __device__ __forceinline__ v16bf load_a_frag(const unsigned short* p, int stride,
                                                    int row0, int col0, int lane) {
  const int r  = lane & 15;
  const int kh = lane >> 4;
  const unsigned short* base = p + (size_t)(row0 + r) * stride + col0 + kh * 16;
  Frag32B f;
  f.a = *(const v4ui*)(base);        // K = kh*16 + 0..7
  f.b = *(const v4ui*)(base + 8);    // K = kh*16 + 8..15
  return __builtin_bit_cast(v16bf, f);
}

// B fragment, transposed source: B[k][n] = src[row0+n][col0+k]
// (for Q*K^T and for weight matrices stored [O][C]).
static __device__ __forceinline__ v16bf load_bT_frag(const unsigned short* p, int stride,
                                                     int row0, int col0, int lane) {
  const int n  = lane & 15;
  const int kh = lane >> 4;
  const unsigned short* base = p + (size_t)(row0 + n) * stride + col0 + kh * 16;
  Frag32B f;
  f.a = *(const v4ui*)(base);        // K = kh*16 + 0..7
  f.b = *(const v4ui*)(base + 8);    // K = kh*16 + 8..15
  return __builtin_bit_cast(v16bf, f);
}

// B fragment, direct source: B[k][n] = src[row0+k][col0+n]  (for P*V)
static __device__ __forceinline__ v16bf load_bD_frag(const unsigned short* p, int stride,
                                                     int row0, int col0, int lane) {
  const int n  = lane & 15;
  const int kh = lane >> 4;
  v16us t;
#pragma unroll
  for (int j = 0; j < 16; ++j)
    t[j] = p[(size_t)(row0 + kh * 16 + j) * stride + col0 + n];
  return __builtin_bit_cast(v16bf, t);
}

static __device__ __forceinline__ v8f wmma_bf16(v16bf a, v16bf b, v8f c) {
  return __builtin_amdgcn_wmma_f32_16x16x32_bf16(false, a, false, b, (short)0, c, false, false);
}

static __device__ __forceinline__ v8f zero_v8f() {
  v8f z;
#pragma unroll
  for (int i = 0; i < 8; ++i) z[i] = 0.0f;
  return z;
}

// ---------------- preprocessing kernels ----------------

__global__ __launch_bounds__(256)
void cvt_weights_kernel(const float* __restrict__ qkvw, const float* __restrict__ projw,
                        unsigned short* __restrict__ wq, unsigned short* __restrict__ wp) {
  int i = blockIdx.x * 256 + threadIdx.x;
  if (i < 3 * CDIM * CDIM) wq[i] = f32_to_bf16(qkvw[i]);
  if (i < CDIM * CDIM)     wp[i] = f32_to_bf16(projw[i]);
}

// fused relative-position bias table: bias_f[h][q][k]
__global__ __launch_bounds__(256)
void build_bias_kernel(const float* __restrict__ table, float* __restrict__ bias_f) {
  int i = blockIdx.x * 256 + threadIdx.x;
  if (i >= NHD * NTOK * NTOK) return;
  int h = i >> 12;
  int q = (i >> 6) & 63;
  int k = i & 63;
  int idx = ((q >> 3) - (k >> 3) + WS - 1) * (2 * WS - 1) + ((q & 7) - (k & 7) + WS - 1);
  bias_f[i] = table[idx * NHD + h];
}

// cyclic shift (-4,-4) + NCHW->NHWC transpose + f32->bf16
__global__ __launch_bounds__(256)
void shift_cvt_kernel(const float* __restrict__ x, unsigned short* __restrict__ xs) {
  const int bh = blockIdx.x;
  const int b  = bh / HH;
  const int hs = bh % HH;
  const int hsrc = (hs + SHIFTSZ) & (HH - 1);
  const int tid = threadIdx.x;

  __shared__ float tile[16][WWID + 1];

  const int wsrc = (tid + SHIFTSZ) & (WWID - 1);
  for (int c0 = 0; c0 < CDIM; c0 += 16) {
    __syncthreads();
#pragma unroll 4
    for (int cc = 0; cc < 16; ++cc)
      tile[cc][tid] = x[(((size_t)b * CDIM + c0 + cc) * HH + hsrc) * WWID + wsrc];
    __syncthreads();
    const int wloc = tid >> 1;
    const int half = tid & 1;
#pragma unroll
    for (int pass = 0; pass < 2; ++pass) {
      int w = pass * 128 + wloc;
      v4ui pv;
#pragma unroll
      for (int i = 0; i < 4; ++i)
        pv[i] = pack2_bf16(tile[half * 8 + 2 * i][w], tile[half * 8 + 2 * i + 1][w]);
      *(v4ui*)&xs[(((size_t)b * HH + hs) * WWID + w) * CDIM + c0 + half * 8] = pv;
    }
  }
}

// ---------------- fused window-attention kernel ----------------
// 1 block per window, 128 threads = 4 wave32 (one wave per head)

#define XS_STRIDE  136                       // bf16 elems, X tile / attn-out tile
#define QS_STRIDE  392                       // bf16 elems, qkv tile [64][384]
#define PS_STRIDE  72                        // bf16 elems, probs [64][64]
#define O2_STRIDE  66                        // f32 elems, out staging [128][64]
#define SM_X_BYTES  (NTOK * XS_STRIDE * 2)               // 17408
#define SM_Q_BYTES  (NTOK * QS_STRIDE * 2)               // 50176
#define SM_P_BYTES  (NHD * NTOK * PS_STRIDE * 2)         // 36864 (>= 128*66*4)
#define SM_BYTES    (SM_X_BYTES + SM_Q_BYTES + SM_P_BYTES)

__global__ __launch_bounds__(128)
void swin_fused_kernel(const unsigned short* __restrict__ xs,
                       const unsigned short* __restrict__ wq,
                       const unsigned short* __restrict__ wp,
                       const float* __restrict__ pb,
                       const float* __restrict__ bias_f,
                       float* __restrict__ out) {
  __shared__ __align__(16) unsigned char smem[SM_BYTES];
  unsigned short* XT  = (unsigned short*)smem;                         // [64][136] X tile; later attn-out
  unsigned short* QKV = (unsigned short*)(smem + SM_X_BYTES);          // [64][392]
  unsigned short* PR  = (unsigned short*)(smem + SM_X_BYTES + SM_Q_BYTES); // [4][64][72] probs
  float*          OT2 = (float*)(smem + SM_X_BYTES + SM_Q_BYTES);      // [128][66] staging (aliases PR)

  const int win = blockIdx.x;
  const int b   = win >> 10;
  const int wh  = (win >> 5) & 31;
  const int ww  = win & 31;
  const int tid  = threadIdx.x;
  const int h    = tid >> 5;                 // wave id == head id
  const int lane = tid & 31;
  const int ln   = lane & 15;
  const int lh   = lane >> 4;

  // ---- 1) load X tile (64 tokens x 128 ch bf16): async global->LDS if available ----
  for (int i = tid; i < NTOK * 16; i += 128) {
    int t = i >> 4, seg = i & 15;
    int tr = t >> 3, tc = t & 7;
    size_t src = (((size_t)b * HH + wh * WS + tr) * WWID + (ww * WS + tc)) * CDIM + seg * 8;
#if __has_builtin(__builtin_amdgcn_global_load_async_to_lds_b128)
    __builtin_amdgcn_global_load_async_to_lds_b128(
        (v4i*)(xs + src),
        (v4i*)&XT[t * XS_STRIDE + seg * 8],
        0, 0);
#else
    *(v4ui*)&XT[t * XS_STRIDE + seg * 8] = *(const v4ui*)&xs[src];
#endif
  }
#if __has_builtin(__builtin_amdgcn_global_load_async_to_lds_b128)
#if __has_builtin(__builtin_amdgcn_s_wait_asynccnt)
  __builtin_amdgcn_s_wait_asynccnt(0);
#else
  asm volatile("s_wait_asynccnt 0" ::: "memory");
#endif
#endif
  __syncthreads();

  // ---- 2) QKV GEMM: wave h computes its head's q,k,v columns ----
#pragma unroll
  for (int m3 = 0; m3 < 3; ++m3) {
#pragma unroll
    for (int j = 0; j < 2; ++j) {
      const int oc0 = m3 * CDIM + h * HDIM + j * 16;  // output channel / row of wq
      v8f acc[4];
#pragma unroll
      for (int mt = 0; mt < 4; ++mt) acc[mt] = zero_v8f();
#pragma unroll
      for (int kc = 0; kc < 4; ++kc) {
        v16bf bf = load_bT_frag(wq, CDIM, oc0, kc * 32, lane);
#pragma unroll
        for (int mt = 0; mt < 4; ++mt) {
          v16bf af = load_a_frag(XT, XS_STRIDE, mt * 16, kc * 32, lane);
          acc[mt] = wmma_bf16(af, bf, acc[mt]);
        }
      }
#pragma unroll
      for (int mt = 0; mt < 4; ++mt)
#pragma unroll
        for (int r = 0; r < 8; r += 2) {
          unsigned int p2 = pack2_bf16(acc[mt][r], acc[mt][r + 1]);
          QKV[(mt * 16 + lh * 8 + r    ) * QS_STRIDE + oc0 + ln] = (unsigned short)p2;
          QKV[(mt * 16 + lh * 8 + r + 1) * QS_STRIDE + oc0 + ln] = (unsigned short)(p2 >> 16);
        }
    }
  }
  __syncthreads();   // XT free after this point (aliased by attn-out)

  // ---- 3) S = Q K^T * scale + bias + mask ----
  const int qc0 = h * HDIM;
  const int kc0 = CDIM + h * HDIM;
  const int vc0 = 2 * CDIM + h * HDIM;
  float sf[4][4][8];
#pragma unroll
  for (int nt = 0; nt < 4; ++nt) {
    v16bf bf = load_bT_frag(QKV, QS_STRIDE, nt * 16, kc0, lane);
#pragma unroll
    for (int mt = 0; mt < 4; ++mt) {
      v16bf af = load_a_frag(QKV, QS_STRIDE, mt * 16, qc0, lane);
      v8f acc = zero_v8f();
      acc = wmma_bf16(af, bf, acc);
#pragma unroll
      for (int r = 0; r < 8; ++r) sf[mt][nt][r] = acc[r];
    }
  }

  const bool wh31 = (wh == NWH - 1);
  const bool ww31 = (ww == NWH - 1);
#pragma unroll
  for (int mt = 0; mt < 4; ++mt)
#pragma unroll
    for (int nt = 0; nt < 4; ++nt)
#pragma unroll
      for (int r = 0; r < 8; ++r) {
        int row = mt * 16 + lh * 8 + r;        // query token
        int col = nt * 16 + ln;                // key token
        int ghq = wh31 ? (((row >> 3) >= SHIFTSZ) ? 2 : 1) : 0;
        int ghk = wh31 ? (((col >> 3) >= SHIFTSZ) ? 2 : 1) : 0;
        int gwq = ww31 ? (((row & 7)  >= SHIFTSZ) ? 2 : 1) : 0;
        int gwk = ww31 ? (((col & 7)  >= SHIFTSZ) ? 2 : 1) : 0;
        float msk = (ghq == ghk && gwq == gwk) ? 0.0f : -100.0f;
        sf[mt][nt][r] = sf[mt][nt][r] * SCALEF + bias_f[(h * NTOK + row) * NTOK + col] + msk;
      }

  // ---- 4) softmax along keys (lane-half shuffle reduction) + bf16 probs ----
  unsigned short* prh = PR + h * NTOK * PS_STRIDE;
#pragma unroll
  for (int mt = 0; mt < 4; ++mt)
#pragma unroll
    for (int r = 0; r < 8; ++r) {
      float mx = -3.0e38f;
#pragma unroll
      for (int nt = 0; nt < 4; ++nt) mx = fmaxf(mx, sf[mt][nt][r]);
#pragma unroll
      for (int d = 1; d < 16; d <<= 1) mx = fmaxf(mx, __shfl_xor(mx, d, 32));
      float sum = 0.0f;
#pragma unroll
      for (int nt = 0; nt < 4; ++nt) {
        float e = __expf(sf[mt][nt][r] - mx);
        sf[mt][nt][r] = e;
        sum += e;
      }
#pragma unroll
      for (int d = 1; d < 16; d <<= 1) sum += __shfl_xor(sum, d, 32);
      float inv = 1.0f / sum;
      int row = mt * 16 + lh * 8 + r;
#pragma unroll
      for (int nt = 0; nt < 4; nt += 2) {
        unsigned int p2 = pack2_bf16(sf[mt][nt][r] * inv, sf[mt][nt + 1][r] * inv);
        prh[row * PS_STRIDE + nt * 16 + ln]        = (unsigned short)p2;
        prh[row * PS_STRIDE + (nt + 1) * 16 + ln]  = (unsigned short)(p2 >> 16);
      }
    }

  // ---- 5) O = P @ V  -> attn-out tile (aliases XT) ----
  {
    v8f of[4][2];
#pragma unroll
    for (int mt = 0; mt < 4; ++mt)
#pragma unroll
      for (int nt = 0; nt < 2; ++nt) of[mt][nt] = zero_v8f();
#pragma unroll
    for (int nt = 0; nt < 2; ++nt)
#pragma unroll
      for (int kc = 0; kc < 2; ++kc) {
        v16bf bf = load_bD_frag(QKV, QS_STRIDE, kc * 32, vc0 + nt * 16, lane);
#pragma unroll
        for (int mt = 0; mt < 4; ++mt) {
          v16bf af = load_a_frag(prh, PS_STRIDE, mt * 16, kc * 32, lane);
          of[mt][nt] = wmma_bf16(af, bf, of[mt][nt]);
        }
      }
#pragma unroll
    for (int mt = 0; mt < 4; ++mt)
#pragma unroll
      for (int nt = 0; nt < 2; ++nt)
#pragma unroll
        for (int r = 0; r < 8; r += 2) {
          unsigned int p2 = pack2_bf16(of[mt][nt][r], of[mt][nt][r + 1]);
          XT[(mt * 16 + lh * 8 + r    ) * XS_STRIDE + h * HDIM + nt * 16 + ln] = (unsigned short)p2;
          XT[(mt * 16 + lh * 8 + r + 1) * XS_STRIDE + h * HDIM + nt * 16 + ln] = (unsigned short)(p2 >> 16);
        }
  }
  __syncthreads();   // attn-out complete; everyone past PR reads (OT2 aliases PR)

  // ---- 6) projection + bias -> staging [channel][token] ----
#pragma unroll
  for (int j2 = 0; j2 < 2; ++j2) {
    const int oc0 = h * HDIM + j2 * 16;
    v8f acc[4];
#pragma unroll
    for (int mt = 0; mt < 4; ++mt) acc[mt] = zero_v8f();
#pragma unroll
    for (int kc = 0; kc < 4; ++kc) {
      v16bf bf = load_bT_frag(wp, CDIM, oc0, kc * 32, lane);
#pragma unroll
      for (int mt = 0; mt < 4; ++mt) {
        v16bf af = load_a_frag(XT, XS_STRIDE, mt * 16, kc * 32, lane);
        acc[mt] = wmma_bf16(af, bf, acc[mt]);
      }
    }
    float pbv = pb[oc0 + ln];
#pragma unroll
    for (int mt = 0; mt < 4; ++mt)
#pragma unroll
      for (int r = 0; r < 8; ++r)
        OT2[(oc0 + ln) * O2_STRIDE + mt * 16 + lh * 8 + r] = acc[mt][r] + pbv;
  }
  __syncthreads();

  // ---- 7) write out with inverse roll (+4,+4); 32B-contiguous token rows ----
  for (int i = tid; i < CDIM * NTOK; i += 128) {
    int o = i >> 6;
    int t = i & 63;
    int ho = (wh * WS + (t >> 3) + SHIFTSZ) & (HH - 1);
    int wo = (ww * WS + (t & 7)  + SHIFTSZ) & (WWID - 1);
    out[(((size_t)b * CDIM + o) * HH + ho) * WWID + wo] = OT2[o * O2_STRIDE + t];
  }
}

// ---------------- launch ----------------
extern "C" void kernel_launch(void* const* d_in, const int* in_sizes, int n_in,
                              void* d_out, int out_size, void* d_ws, size_t ws_size,
                              hipStream_t stream) {
  (void)in_sizes; (void)n_in; (void)out_size; (void)ws_size;
  const float* x     = (const float*)d_in[0];
  const float* qkvw  = (const float*)d_in[1];
  const float* projw = (const float*)d_in[2];
  const float* projb = (const float*)d_in[3];
  const float* btab  = (const float*)d_in[4];
  float* out = (float*)d_out;

  unsigned short* xs = (unsigned short*)d_ws;                       // [B,H,W,C] bf16 shifted
  unsigned short* wq = xs + (size_t)BATCH * HH * WWID * CDIM;       // [384,128] bf16
  unsigned short* wp = wq + 3 * CDIM * CDIM;                        // [128,128] bf16
  float* bias_f = (float*)(wp + CDIM * CDIM);                       // [4,64,64] f32

  cvt_weights_kernel<<<(3 * CDIM * CDIM + 255) / 256, 256, 0, stream>>>(qkvw, projw, wq, wp);
  build_bias_kernel<<<(NHD * NTOK * NTOK + 255) / 256, 256, 0, stream>>>(btab, bias_f);
  shift_cvt_kernel<<<BATCH * HH, 256, 0, stream>>>(x, xs);
  swin_fused_kernel<<<NWIN, 128, 0, stream>>>(xs, wq, wp, projb, bias_f, out);
}
